// WindowAttention2D_87359634801295
// MI455X (gfx1250) — matrix operand order repaired and checked by reading, not compiled, not run on hardware
//
#include <hip/hip_runtime.h>
#include <hip/hip_bf16.h>
#include <math.h>

typedef __attribute__((ext_vector_type(16))) _Float16 v16h;
typedef __attribute__((ext_vector_type(8)))  _Float16 v8h;
typedef __attribute__((ext_vector_type(8)))  float    v8f;

#define WMMA_F16(a,b,c) __builtin_amdgcn_wmma_f32_16x16x32_f16(false,(a),false,(b),(short)0,(c),false,false)

// ---------------- WMMA operand loaders (per CDNA5 ISA 7.12.2 layouts) ----------------

// A matrix 16x32 f16, row-major stride ld (halves).
// lanes 0-15: halves 0-7 -> K0-7, halves 8-15 -> K16-23
// lanes 16-31: halves 0-7 -> K8-15, halves 8-15 -> K24-31
__device__ __forceinline__ v16h load_a_lds(const _Float16* base, int ld, int m0, int k0, int lane) {
  int m   = m0 + (lane & 15);
  int off = m * ld + k0 + 8 * (lane >> 4);
  union { v16h v; v8h h[2]; } u;
  u.h[0] = *(const v8h*)(base + off);
  u.h[1] = *(const v8h*)(base + off + 16);
  return u.v;
}

// B matrix 32x16 f16 where memory is [N][K] row-major (weights [out][in], k, vT).
// B[k][n] = src[(n0+n)*ld + k]; lane: column n=lane&15, halves 0-7 -> K k0+8*(lane>=16)+i, halves 8-15 -> K k0+16+8*(lane>=16)+(i-8)
__device__ __forceinline__ v16h load_b_nk(const _Float16* base, int ld, int n0, int k0, int lane) {
  int n = n0 + (lane & 15);
  int k = k0 + 8 * (lane >> 4);
  const _Float16* p = base + n * ld + k;
  union { v16h v; v8h h[2]; } u;
  u.h[0] = *(const v8h*)(p);
  u.h[1] = *(const v8h*)(p + 16);
  return u.v;
}

// ---------------- weight f32 -> f16 prep ----------------
// ws layout (halves): qkv[196608] | proj[65536] | fc1[262144] | fc2[262144]
__global__ void convert_weights_kernel(const float* __restrict__ qkv_w,
                                       const float* __restrict__ proj_w,
                                       const float* __restrict__ fc1_w,
                                       const float* __restrict__ fc2_w,
                                       _Float16* __restrict__ ws) {
  int i = blockIdx.x * blockDim.x + threadIdx.x;   // 0 .. 262143
  if (i < 196608) ws[i]          = (_Float16)qkv_w[i];
  if (i < 65536)  ws[196608 + i] = (_Float16)proj_w[i];
  if (i < 262144) ws[262144 + i] = (_Float16)fc1_w[i];
  if (i < 262144) ws[524288 + i] = (_Float16)fc2_w[i];
}

// ---------------- main: one workgroup (8 wave32) per 8x8 window ----------------
// LDS (dynamic):
//   tok   f32 [64][257]      @ 0        (65792 B)  residual stream
//   abuf  f16 [64][264]      @ 65792    (33792 B)  LN out / attn out (reused)
//   qkv16 f16 [64][768]      @ 99584    (98304 B)  q|k rows; MLP hidden reuse
//   vT    f16 [256][72]      @ 197888   (36864 B)  V transposed: vT[h*32+d][s]
//   at16  f16 8x[64][72]     @ 234752   (73728 B)  per-wave softmaxed attn
// total 308480 B  (<= 320KB WGP limit)
#define SMEM_BYTES 308480

__global__ __launch_bounds__(256) void win_block_kernel(
    const float* __restrict__ x,
    const float* __restrict__ qkv_b,
    const float* __restrict__ proj_b,
    const float* __restrict__ n1_g, const float* __restrict__ n1_b,
    const float* __restrict__ n2_g, const float* __restrict__ n2_b,
    const float* __restrict__ fc1_b,
    const float* __restrict__ fc2_b,
    const _Float16* __restrict__ w_qkv,    // [768][256]
    const _Float16* __restrict__ w_proj,   // [256][256]
    const _Float16* __restrict__ w_fc1,    // [1024][256]
    const _Float16* __restrict__ w_fc2,    // [256][1024]
    float* __restrict__ out) {
  extern __shared__ char smem[];
  float*    tok   = (float*)(smem);
  _Float16* abuf  = (_Float16*)(smem + 65792);
  _Float16* qkv16 = (_Float16*)(smem + 99584);
  _Float16* vT    = (_Float16*)(smem + 197888);
  _Float16* at16  = (_Float16*)(smem + 234752);
  _Float16* hbuf  = qkv16;   // MLP hidden chunk reuses qkv region

  const int tid  = threadIdx.x;
  const int lane = tid & 31;
  // wave id, forced into an SGPR (wave-uniform) so per-wave control flow is scalar
  const int wvs  = __builtin_amdgcn_readfirstlane(tid >> 5);   // 0..7
  const int widx = blockIdx.x;        // 0..4095
  const int b  = widx >> 8;
  const int wh = (widx >> 4) & 15;
  const int ww = widx & 15;
  const int xbase = ((b * 256) * 128 + wh * 8) * 128 + ww * 8;  // + c*16384 + ty*128 + tx

  // per-wave fixed M-tile; N-range split across wave pairs
  const int m0   = (wvs & 3) * 16;
  const int whi  = wvs >> 2;           // 0 or 1
  const int ncol = lane & 15;
  const int mrow = 8 * (lane >> 4);

  // ---- load window tokens: tok[t][c] = x[b][c][wh*8+ty][ww*8+tx] ----
  {
    int t = tid & 63, cq = tid >> 6;
    int ty = t >> 3, tx = t & 7;
    int base = xbase + ty * 128 + tx;
    for (int cc = 0; cc < 64; ++cc) {
      int c = cq * 64 + cc;
      tok[t * 257 + c] = x[base + c * 16384];
    }
  }
  __syncthreads();

  // ---- layernorm: tok -> abuf (f16); 4 lanes per row, shfl reduction ----
  auto layernorm = [&](const float* g, const float* bb) {
    int r = tid >> 2, q = tid & 3;
    const float* row = tok + r * 257 + q * 64;
    float s = 0.f;
    for (int c = 0; c < 64; ++c) s += row[c];
    s += __shfl_xor(s, 1, 32);
    s += __shfl_xor(s, 2, 32);
    float mu = s * (1.f / 256.f);
    float v = 0.f;
    for (int c = 0; c < 64; ++c) { float d = row[c] - mu; v += d * d; }
    v += __shfl_xor(v, 1, 32);
    v += __shfl_xor(v, 2, 32);
    float rs = rsqrtf(v * (1.f / 256.f) + 1e-5f);
    _Float16* o = abuf + r * 264 + q * 64;
    const float* gg = g + q * 64;
    const float* bg = bb + q * 64;
    for (int c = 0; c < 64; ++c)
      o[c] = (_Float16)((row[c] - mu) * rs * gg[c] + bg[c]);
  };

  layernorm(n1_g, n1_b);
  __syncthreads();

  // ==== QKV GEMM: [64x256] x [256x768] -> qkv16 (q,k) / vT (v, transposed) ====
  {
    v16h afr[8];
#pragma unroll
    for (int ks = 0; ks < 8; ++ks) afr[ks] = load_a_lds(abuf, 264, m0, ks * 32, lane);

    // wave half 0: nt 0..23 (all q/k).  wave half 1: nt 24..31 q/k, nt 32..47 v.
    const int nqk = whi ? 8 : 24;       // scalar per wave
    for (int t = 0; t < 24; ++t) {
      int nt = whi * 24 + t;
      int n0 = nt * 16;
      v8f acc = {};
      v16h bc = load_b_nk(w_qkv, 256, n0, 0, lane);
#pragma unroll
      for (int ks = 0; ks < 8; ++ks) {
        v16h bn = (ks < 7) ? load_b_nk(w_qkv, 256, n0, (ks + 1) * 32, lane) : bc;
        acc = WMMA_F16(afr[ks], bc, acc);
        bc = bn;
      }
      int n = n0 + ncol;
      float bias = qkv_b[n];
      int mb = m0 + mrow;
      if (t < nqk) {            // q,k thirds: row-major [t][o]   (scalar branch)
#pragma unroll
        for (int j = 0; j < 8; ++j)
          qkv16[(mb + j) * 768 + n] = (_Float16)(acc[j] + bias);
      } else {                  // v third: transposed  vT[h*32+d][s]
        int dp = n - 512;
#pragma unroll
        for (int j = 0; j < 8; ++j)
          vT[dp * 72 + (mb + j)] = (_Float16)(acc[j] + bias);
      }
    }
  }
  __syncthreads();

  // ==== attention: one wave owns one head; softmax in registers; no barriers ====
  {
    const int h = wvs;
    const _Float16* qb = qkv16 + h * 32;          // q[t][d] stride 768
    const _Float16* kb = qkv16 + 256 + h * 32;    // k[s][d] stride 768
    const _Float16* vb = vT + (h * 32) * 72;      // vT[d][s] stride 72
    _Float16* myat = at16 + h * 4608;             // [64][72] wave-private

    // k fragments for all 4 N-tiles (K=32, single step)
    v16h bk0 = load_b_nk(kb, 768, 0,  0, lane);
    v16h bk1 = load_b_nk(kb, 768, 16, 0, lane);
    v16h bk2 = load_b_nk(kb, 768, 32, 0, lane);
    v16h bk3 = load_b_nk(kb, 768, 48, 0, lane);

    const float scale = 0.17677669529663689f;     // 1/sqrt(32)
#pragma unroll
    for (int mtl = 0; mtl < 4; ++mtl) {
      v16h aq = load_a_lds(qb, 768, mtl * 16, 0, lane);
      v8f s0 = {}, s1 = {}, s2 = {}, s3 = {};
      s0 = WMMA_F16(aq, bk0, s0);
      s1 = WMMA_F16(aq, bk1, s1);
      s2 = WMMA_F16(aq, bk2, s2);
      s3 = WMMA_F16(aq, bk3, s3);
      int mb = mtl * 16 + mrow;
#pragma unroll
      for (int j = 0; j < 8; ++j) {
        // row m = mb+j lives across 16 lanes (one half) x 4 tiles
        float v0 = s0[j] * scale, v1 = s1[j] * scale;
        float v2 = s2[j] * scale, v3 = s3[j] * scale;
        float mx = fmaxf(fmaxf(v0, v1), fmaxf(v2, v3));
        mx = fmaxf(mx, __shfl_xor(mx, 1, 32));
        mx = fmaxf(mx, __shfl_xor(mx, 2, 32));
        mx = fmaxf(mx, __shfl_xor(mx, 4, 32));
        mx = fmaxf(mx, __shfl_xor(mx, 8, 32));
        float e0 = __expf(v0 - mx), e1 = __expf(v1 - mx);
        float e2 = __expf(v2 - mx), e3 = __expf(v3 - mx);
        float sum = (e0 + e1) + (e2 + e3);
        sum += __shfl_xor(sum, 1, 32);
        sum += __shfl_xor(sum, 2, 32);
        sum += __shfl_xor(sum, 4, 32);
        sum += __shfl_xor(sum, 8, 32);
        float inv = 1.f / sum;
        _Float16* rp = myat + (mb + j) * 72 + ncol;
        rp[0]  = (_Float16)(e0 * inv);
        rp[16] = (_Float16)(e1 * inv);
        rp[32] = (_Float16)(e2 * inv);
        rp[48] = (_Float16)(e3 * inv);
      }
    }

    // o = attn @ v : all 8 output tiles of this head (K=64, 2 steps)
    v16h bv00 = load_b_nk(vb, 72, 0,  0,  lane);
    v16h bv01 = load_b_nk(vb, 72, 0,  32, lane);
    v16h bv10 = load_b_nk(vb, 72, 16, 0,  lane);
    v16h bv11 = load_b_nk(vb, 72, 16, 32, lane);
#pragma unroll
    for (int mtl = 0; mtl < 4; ++mtl) {
      v16h a0 = load_a_lds(myat, 72, mtl * 16, 0,  lane);
      v16h a1 = load_a_lds(myat, 72, mtl * 16, 32, lane);
      int mb = mtl * 16 + mrow;
      v8f acc = {};
      acc = WMMA_F16(a0, bv00, acc);
      acc = WMMA_F16(a1, bv01, acc);
      int n = h * 32 + ncol;
#pragma unroll
      for (int j = 0; j < 8; ++j) abuf[(mb + j) * 264 + n] = (_Float16)acc[j];
      v8f acd = {};
      acd = WMMA_F16(a0, bv10, acd);
      acd = WMMA_F16(a1, bv11, acd);
#pragma unroll
      for (int j = 0; j < 8; ++j) abuf[(mb + j) * 264 + n + 16] = (_Float16)acd[j];
    }
  }
  __syncthreads();

  // ==== proj GEMM: [64x256] x [256x256], residual into tok ====
  {
    v16h afr[8];
#pragma unroll
    for (int ks = 0; ks < 8; ++ks) afr[ks] = load_a_lds(abuf, 264, m0, ks * 32, lane);
    for (int t = 0; t < 8; ++t) {
      int n0 = (whi * 8 + t) * 16;
      v8f acc = {};
      v16h bc = load_b_nk(w_proj, 256, n0, 0, lane);
#pragma unroll
      for (int ks = 0; ks < 8; ++ks) {
        v16h bn = (ks < 7) ? load_b_nk(w_proj, 256, n0, (ks + 1) * 32, lane) : bc;
        acc = WMMA_F16(afr[ks], bc, acc);
        bc = bn;
      }
      int n = n0 + ncol;
      float bias = proj_b[n];
      int mb = m0 + mrow;
#pragma unroll
      for (int j = 0; j < 8; ++j)
        tok[(mb + j) * 257 + n] += acc[j] + bias;
    }
  }
  __syncthreads();

  layernorm(n2_g, n2_b);
  __syncthreads();

  // ==== MLP: fc1(256->1024) + exact GELU + fc2(1024->256), hidden chunked by 256 ====
  v8f acc2[4][2];
#pragma unroll
  for (int m = 0; m < 4; ++m)
#pragma unroll
    for (int i = 0; i < 2; ++i) acc2[m][i] = (v8f){};

  // preload LN2 A fragments once (abuf is stable through all chunks)
  v16h afr2[8];
#pragma unroll
  for (int ks = 0; ks < 8; ++ks) afr2[ks] = load_a_lds(abuf, 264, m0, ks * 32, lane);

  for (int ch = 0; ch < 4; ++ch) {
    // hbuf = gelu(y @ fc1_chunk + b1)  [64 x 256]
    const _Float16* wf1 = w_fc1 + (ch * 256) * 256;
    for (int t = 0; t < 8; ++t) {
      int n0 = (whi * 8 + t) * 16;
      v8f acc = {};
      v16h bc = load_b_nk(wf1, 256, n0, 0, lane);
#pragma unroll
      for (int ks = 0; ks < 8; ++ks) {
        v16h bn = (ks < 7) ? load_b_nk(wf1, 256, n0, (ks + 1) * 32, lane) : bc;
        acc = WMMA_F16(afr2[ks], bc, acc);
        bc = bn;
      }
      int n = n0 + ncol;
      float bias = fc1_b[ch * 256 + n];
      int mb = m0 + mrow;
#pragma unroll
      for (int j = 0; j < 8; ++j) {
        float v = acc[j] + bias;
        v = 0.5f * v * (1.f + erff(v * 0.7071067811865475f));   // exact gelu
        hbuf[(mb + j) * 264 + n] = (_Float16)v;
      }
    }
    __syncthreads();

    // acc2 += hbuf @ fc2_chunk ; persistent register accumulators (8 tiles/wave)
#pragma unroll
    for (int m = 0; m < 4; ++m) {
      v16h ah[8];
#pragma unroll
      for (int ks = 0; ks < 8; ++ks) ah[ks] = load_a_lds(hbuf, 264, m * 16, ks * 32, lane);
#pragma unroll
      for (int i = 0; i < 2; ++i) {
        int n0 = (wvs * 2 + i) * 16;
        v16h bc = load_b_nk(w_fc2, 1024, n0, ch * 256, lane);
#pragma unroll
        for (int ks = 0; ks < 8; ++ks) {
          v16h bn = (ks < 7) ? load_b_nk(w_fc2, 1024, n0, ch * 256 + (ks + 1) * 32, lane) : bc;
          acc2[m][i] = WMMA_F16(ah[ks], bc, acc2[m][i]);
          bc = bn;
        }
      }
    }
    __syncthreads();
  }

  // ==== epilogue: tok += mlp + b2, then coalesced store back to [B,C,H,W] ====
#pragma unroll
  for (int m = 0; m < 4; ++m) {
#pragma unroll
    for (int i = 0; i < 2; ++i) {
      int n  = (wvs * 2 + i) * 16 + ncol;
      float bias = fc2_b[n];
      int mb = m * 16 + mrow;
#pragma unroll
      for (int j = 0; j < 8; ++j)
        tok[(mb + j) * 257 + n] += acc2[m][i][j] + bias;
    }
  }
  __syncthreads();

  {
    int t = tid & 63, cq = tid >> 6;
    int ty = t >> 3, tx = t & 7;
    int base = xbase + ty * 128 + tx;
    for (int cc = 0; cc < 64; ++cc) {
      int c = cq * 64 + cc;
      out[base + c * 16384] = tok[t * 257 + c];
    }
  }
}

extern "C" void kernel_launch(void* const* d_in, const int* in_sizes, int n_in,
                              void* d_out, int out_size, void* d_ws, size_t ws_size,
                              hipStream_t stream) {
  const float* x      = (const float*)d_in[0];
  const float* qkv_w  = (const float*)d_in[1];
  const float* qkv_b  = (const float*)d_in[2];
  const float* proj_w = (const float*)d_in[3];
  const float* proj_b = (const float*)d_in[4];
  const float* n1_g   = (const float*)d_in[5];
  const float* n1_b   = (const float*)d_in[6];
  const float* n2_g   = (const float*)d_in[7];
  const float* n2_b   = (const float*)d_in[8];
  const float* fc1_w  = (const float*)d_in[9];
  const float* fc1_b  = (const float*)d_in[10];
  const float* fc2_w  = (const float*)d_in[11];
  const float* fc2_b  = (const float*)d_in[12];
  _Float16* w16 = (_Float16*)d_ws;

  convert_weights_kernel<<<1024, 256, 0, stream>>>(qkv_w, proj_w, fc1_w, fc2_w, w16);

  hipFuncSetAttribute(reinterpret_cast<const void*>(&win_block_kernel),
                      hipFuncAttributeMaxDynamicSharedMemorySize, SMEM_BYTES);

  win_block_kernel<<<4096, 256, SMEM_BYTES, stream>>>(
      x, qkv_b, proj_b, n1_g, n1_b, n2_g, n2_b, fc1_b, fc2_b,
      w16,                 // qkv  [768][256]
      w16 + 196608,        // proj [256][256]
      w16 + 262144,        // fc1  [1024][256]
      w16 + 524288,        // fc2  [256][1024]
      (float*)d_out);
}
